// AggregateGCN_71932112273949
// MI455X (gfx1250) — matrix operand order repaired and checked by reading, not compiled, not run on hardware
//
#include <hip/hip_runtime.h>
#include <hip/hip_bf16.h>

// ---- problem constants (match reference) ----
#define TOTAL_N   131072      // B*N nodes
#define NPG       2048        // nodes per graph
#define NGRAPH    64
#define IN_DIM    128
#define HID       256
#define EMB       64

typedef __attribute__((ext_vector_type(16))) _Float16 v16h;
typedef __attribute__((ext_vector_type(8)))  _Float16 h8;
typedef __attribute__((ext_vector_type(8)))  float    v8f;
typedef __attribute__((ext_vector_type(4)))  unsigned int u32x4;
typedef __attribute__((ext_vector_type(8)))  int      i32x8;
typedef __attribute__((ext_vector_type(4)))  int      i32x4;

union Frag16 { v16h v; h8 h[2]; };

// -------------------------------------------------------------------------
// Degree accumulation: one thread per edge.
// -------------------------------------------------------------------------
__global__ void degree_kernel(const int* __restrict__ src, const int* __restrict__ dst,
                              float* __restrict__ outd, float* __restrict__ ind, int E) {
    int e = blockIdx.x * blockDim.x + threadIdx.x;
    if (e < E) {
        atomicAdd(&outd[src[e]], 1.0f);
        atomicAdd(&ind[dst[e]], 1.0f);
    }
}

// deg -> max(deg,1)^-0.5 in place (outd and ind contiguous, one launch)
__global__ void norm_kernel(float* __restrict__ d, int n) {
    int i = blockIdx.x * blockDim.x + threadIdx.x;
    if (i < n) d[i] = rsqrtf(fmaxf(d[i], 1.0f));
}

// -------------------------------------------------------------------------
// Weight prep: fp32 row-major W[K][N] -> f16 N-major Wt[N][K].
// Makes every WMMA B-fragment two contiguous 16-byte runs per lane.
// -------------------------------------------------------------------------
__global__ void wprep_kernel(const float* __restrict__ W, _Float16* __restrict__ Wt,
                             int K, int N) {
    int i = blockIdx.x * blockDim.x + threadIdx.x;
    if (i < K * N) {
        int n = i / K;
        int k = i - n * K;
        Wt[i] = (_Float16)W[(size_t)k * N + n];
    }
}

// -------------------------------------------------------------------------
// Edge scatter (SpMM): agg[dst] += h[src] * out_norm[src]
// 64 threads per edge, float4 per thread (256 features). Memory-bound path:
// h (128MB) is mostly L2-resident on MI455X (192MB global L2).
// -------------------------------------------------------------------------
__global__ void edge_scatter_kernel(const float* __restrict__ h,
                                    const float* __restrict__ outnorm,
                                    const int* __restrict__ src,
                                    const int* __restrict__ dst,
                                    float* __restrict__ agg, int E) {
    int t = threadIdx.x;                      // 256 threads
    long e = (long)blockIdx.x * 4 + (t >> 6); // 4 edges per block
    int f = (t & 63) << 2;                    // feature float4 offset
    if (e >= E) return;
    int s = src[e], d = dst[e];
    float sc = outnorm[s];
    const float4 v = *(const float4*)(h + (size_t)s * HID + f);
    float* ap = agg + (size_t)d * HID + f;
    atomicAdd(ap + 0, v.x * sc);
    atomicAdd(ap + 1, v.y * sc);
    atomicAdd(ap + 2, v.z * sc);
    atomicAdd(ap + 3, v.w * sc);
}

// -------------------------------------------------------------------------
// TDM descriptor issue: DMA a 2D fp32 tile (rows x 32, row stride K elems)
// from global memory into LDS. ISA 08_async_tensor 8.3/8.4 field packing.
// Toolchain uses the 6-arg builtin form (extra zero group before cpol).
// -------------------------------------------------------------------------
__device__ __forceinline__ void tdm_load_tile_f32(const float* gsrc, void* ldsDst,
                                                  int rows, int K) {
    unsigned long long gaddr = (unsigned long long)(uintptr_t)gsrc;
    unsigned int ldsAddr = (unsigned int)(uintptr_t)ldsDst;   // low 32b = LDS offset

    // group0: count=1 | lds_addr | global_addr[56:0] | type=2
    u32x4 g0;
    g0[0] = 1u;                                   // count=1, user descriptor
    g0[1] = ldsAddr;
    g0[2] = (unsigned int)(gaddr & 0xFFFFFFFFu);
    g0[3] = (unsigned int)((gaddr >> 32) & 0x1FFFFFFu) | (2u << 30);  // type=2

    // group1: mask=0, data_size=2 (4B), tensor_dim0=K, tensor_dim1=big,
    //         tile_dim0=32, tile_dim1=rows, tile_dim2=0, dim0_stride=K
    unsigned int td0 = (unsigned int)K;
    unsigned int td1 = 1u << 20;
    unsigned long long st0 = (unsigned long long)K;
    i32x8 g1;
    g1[0] = (int)(2u << 16);                                   // data_size=4B
    g1[1] = (int)((td0 & 0xFFFFu) << 16);                      // tensor_dim0 lo
    g1[2] = (int)(((td0 >> 16) & 0xFFFFu) | ((td1 & 0xFFFFu) << 16));
    g1[3] = (int)(((td1 >> 16) & 0xFFFFu) | (32u << 16));      // tile_dim0=32
    g1[4] = (int)((unsigned int)rows & 0xFFFFu);               // tile_dim1, tile_dim2=0
    g1[5] = (int)(unsigned int)(st0 & 0xFFFFFFFFu);            // dim0_stride lo
    g1[6] = (int)(unsigned int)((st0 >> 32) & 0xFFFFu);        // dim0_stride hi, dim1_stride=0
    g1[7] = 0;

    i32x4 gz4 = { 0, 0, 0, 0 };                                // 2D: groups 2/3 unused
    i32x8 gz8 = { 0, 0, 0, 0, 0, 0, 0, 0 };
    __builtin_amdgcn_tensor_load_to_lds(g0, g1, gz4, gz4, gz8, 0);
}

// -------------------------------------------------------------------------
// WMMA GEMM:  C[M,N] = act( (A .* rownorm) @ W + bias )
//
// Per K-chunk: wave0 issues one TDM tensor_load_to_lds for the fp32 A tile
// (rowsPerBlock x 32), waits TENSORcnt==0, barrier. Each wave builds its A
// fragment from LDS (4x ds_load_b128, one row per lane) with rownorm fold +
// f16 convert, and B fragments as two contiguous 16B loads per lane from the
// pre-transposed f16 Wt. 4 WMMAs per chunk per wave (16x64 tile).
//
// CDNA5 fragment layouts (ISA 7.12.2, wave32):
//  A (16x32 f16): lane row = lane&15; elems 0..7 <-> K=base..base+7,
//                 elems 8..15 <-> K=base+16..base+23, base = 8*(lane>=16).
//  B (32x16 f16): col=lane&15; elems 0..7 <-> K=kB..kB+7, elems 8..15 <-> K=kB+16..kB+23, kB=8*(lane>=16).
//  C/D (16x16 f32): VGPR i -> row = i + 8*(lane>=16), col = lane&15.
// -------------------------------------------------------------------------
__global__ void gemm_wmma_kernel(const float* __restrict__ A,
                                 const float* __restrict__ rownorm,   // may be null
                                 const _Float16* __restrict__ Wt,     // [Nout][K] f16
                                 const float* __restrict__ bias,
                                 float* __restrict__ C,
                                 int M, int K, int Nout, int doRelu) {
    __shared__ float Aldsf[64 * 32];            // up to 64 rows x 32 K fp32 (8KB)

    const int lane    = threadIdx.x & 31;
    const int wave    = threadIdx.x >> 5;
    const int wavesPerBlock = blockDim.x >> 5;
    const int nTiles  = Nout >> 6;              // 64-wide N tiles per row strip
    const int mSub    = wave / nTiles;
    const int nTile   = wave - mSub * nTiles;
    const int rowsPerBlock = (wavesPerBlock / nTiles) << 4;
    const int rowBase = blockIdx.x * rowsPerBlock;
    const int nBase   = nTile << 6;

    const int r       = lane & 15;
    const int halfSel = lane >> 4;
    const int rowLocal = (mSub << 4) + r;       // A-fragment row for this lane
    const float rn = rownorm ? rownorm[rowBase + rowLocal] : 1.0f;

    v8f acc[4] = {};

    for (int k0 = 0; k0 < K; k0 += 32) {
        __syncthreads();                         // prior chunk fully consumed
        if (wave == 0) {                         // one TDM issue per block
            tdm_load_tile_f32(A + (size_t)rowBase * K + k0, Aldsf, rowsPerBlock, K);
            __builtin_amdgcn_s_wait_tensorcnt(0);
        }
        __syncthreads();                         // LDS tile visible to all waves

        // ---- A fragment: 4x ds_load_b128 (one row per lane) + f16 convert ----
        const int aBase = halfSel << 3;
        const float* Ar = &Aldsf[rowLocal << 5];
        const float4 q0 = *(const float4*)(Ar + aBase);
        const float4 q1 = *(const float4*)(Ar + aBase + 4);
        const float4 q2 = *(const float4*)(Ar + aBase + 16);
        const float4 q3 = *(const float4*)(Ar + aBase + 20);
        v16h a;
        a[0]  = (_Float16)(q0.x * rn);  a[1]  = (_Float16)(q0.y * rn);
        a[2]  = (_Float16)(q0.z * rn);  a[3]  = (_Float16)(q0.w * rn);
        a[4]  = (_Float16)(q1.x * rn);  a[5]  = (_Float16)(q1.y * rn);
        a[6]  = (_Float16)(q1.z * rn);  a[7]  = (_Float16)(q1.w * rn);
        a[8]  = (_Float16)(q2.x * rn);  a[9]  = (_Float16)(q2.y * rn);
        a[10] = (_Float16)(q2.z * rn);  a[11] = (_Float16)(q2.w * rn);
        a[12] = (_Float16)(q3.x * rn);  a[13] = (_Float16)(q3.y * rn);
        a[14] = (_Float16)(q3.z * rn);  a[15] = (_Float16)(q3.w * rn);

        // ---- B fragments: two 16B runs per lane (K=kB..+7, kB+16..+23) from transposed f16 W ----
        const int kB = k0 + (halfSel << 3);
        #pragma unroll
        for (int sub = 0; sub < 4; ++sub) {
            const int col = nBase + (sub << 4) + r;
            const _Float16* wp = Wt + (size_t)col * K + kB;
            Frag16 bf;
            bf.h[0] = *(const h8*)(wp);
            bf.h[1] = *(const h8*)(wp + 16);
            acc[sub] = __builtin_amdgcn_wmma_f32_16x16x32_f16(
                false, a, false, bf.v, (short)0, acc[sub], false, false);
        }
    }

    // ---- epilogue: bias + optional ReLU, C/D layout store ----
    #pragma unroll
    for (int sub = 0; sub < 4; ++sub) {
        const int col = nBase + (sub << 4) + r;
        const float bv = bias[col];
        #pragma unroll
        for (int i = 0; i < 8; ++i) {
            const int orow = rowBase + (mSub << 4) + (halfSel << 3) + i;
            float v = acc[sub][i] + bv;
            if (doRelu) v = fmaxf(v, 0.0f);
            C[(size_t)orow * Nout + col] = v;
        }
    }
}

// -------------------------------------------------------------------------
// Gather agent rows (node 0 of each graph) into a dense [64, HID] buffer.
// -------------------------------------------------------------------------
__global__ void agent_gather_kernel(const float* __restrict__ h, float* __restrict__ agent) {
    int i = blockIdx.x * blockDim.x + threadIdx.x;   // NGRAPH*HID
    if (i < NGRAPH * HID) {
        int g = i >> 8;          // /HID
        int f = i & (HID - 1);
        agent[i] = h[((size_t)g * NPG) * HID + f];
    }
}

// -------------------------------------------------------------------------
static inline void launch_gemm(const float* A, const float* rn, const _Float16* Wt,
                               const float* b, float* C, int M, int K, int Nout,
                               int relu, hipStream_t s) {
    int threads = (Nout >= 256) ? 256 : 128;      // 8 or 4 waves
    int wavesPerBlock = threads >> 5;
    int nTiles = Nout >> 6;
    int rowsPerBlock = (wavesPerBlock / nTiles) << 4;
    int blocks = M / rowsPerBlock;
    gemm_wmma_kernel<<<blocks, threads, 0, s>>>(A, rn, Wt, b, C, M, K, Nout, relu);
}

extern "C" void kernel_launch(void* const* d_in, const int* in_sizes, int n_in,
                              void* d_out, int out_size, void* d_ws, size_t ws_size,
                              hipStream_t stream) {
    const float* x     = (const float*)d_in[0];
    const int*   src   = (const int*)d_in[1];
    const int*   dst   = (const int*)d_in[2];
    //            d_in[3] = num_nodes, d_in[4] = nodes_per_graph (compile-time here)
    const float* w_lin = (const float*)d_in[5];
    const float* b_lin = (const float*)d_in[6];
    const float* w_c0  = (const float*)d_in[7];
    const float* b_c0  = (const float*)d_in[8];
    const float* w_c1  = (const float*)d_in[9];
    const float* b_c1  = (const float*)d_in[10];
    const float* w_emb = (const float*)d_in[11];
    const float* b_emb = (const float*)d_in[12];
    const int E = in_sizes[1];

    // ---- workspace layout ----
    float* ws      = (float*)d_ws;
    float* outnorm = ws;                                     // TOTAL_N
    float* innorm  = ws + TOTAL_N;                           // TOTAL_N
    float* hA      = ws + 2 * (size_t)TOTAL_N;               // TOTAL_N*HID
    float* aggB    = hA + (size_t)TOTAL_N * HID;             // TOTAL_N*HID
    float* agent   = aggB + (size_t)TOTAL_N * HID;           // NGRAPH*HID
    _Float16* wt_lin = (_Float16*)(agent + NGRAPH * HID);    // f16 region
    _Float16* wt_c0  = wt_lin + IN_DIM * HID;
    _Float16* wt_c1  = wt_c0 + HID * HID;
    _Float16* wt_emb = wt_c1 + HID * HID;

    // ---- weight transpose+convert (tiny, once per call) ----
    wprep_kernel<<<(IN_DIM * HID + 255) / 256, 256, 0, stream>>>(w_lin, wt_lin, IN_DIM, HID);
    wprep_kernel<<<(HID * HID + 255) / 256, 256, 0, stream>>>(w_c0, wt_c0, HID, HID);
    wprep_kernel<<<(HID * HID + 255) / 256, 256, 0, stream>>>(w_c1, wt_c1, HID, HID);
    wprep_kernel<<<(HID * EMB + 255) / 256, 256, 0, stream>>>(w_emb, wt_emb, HID, EMB);

    // ---- degrees -> symmetric norms ----
    (void)hipMemsetAsync(outnorm, 0, 2 * (size_t)TOTAL_N * sizeof(float), stream);
    degree_kernel<<<(E + 255) / 256, 256, 0, stream>>>(src, dst, outnorm, innorm, E);
    norm_kernel<<<(2 * TOTAL_N + 255) / 256, 256, 0, stream>>>(outnorm, 2 * TOTAL_N);

    // ---- h1 = relu(x @ w_lin + b_lin) ----
    launch_gemm(x, nullptr, wt_lin, b_lin, hA, TOTAL_N, IN_DIM, HID, 1, stream);

    // ---- conv0: agg = scatter(h1*outnorm); h2 = relu((agg*innorm) @ w_c0 + b_c0) ----
    (void)hipMemsetAsync(aggB, 0, (size_t)TOTAL_N * HID * sizeof(float), stream);
    edge_scatter_kernel<<<(E + 3) / 4, 256, 0, stream>>>(hA, outnorm, src, dst, aggB, E);
    launch_gemm(aggB, innorm, wt_c0, b_c0, hA, TOTAL_N, HID, HID, 1, stream);

    // ---- conv1 ----
    (void)hipMemsetAsync(aggB, 0, (size_t)TOTAL_N * HID * sizeof(float), stream);
    edge_scatter_kernel<<<(E + 3) / 4, 256, 0, stream>>>(hA, outnorm, src, dst, aggB, E);
    launch_gemm(aggB, innorm, wt_c1, b_c1, hA, TOTAL_N, HID, HID, 1, stream);

    // ---- head: agent rows @ w_emb + b_emb -> d_out [64, 64] ----
    agent_gather_kernel<<<(NGRAPH * HID + 255) / 256, 256, 0, stream>>>(hA, agent);
    launch_gemm(agent, nullptr, wt_emb, b_emb, (float*)d_out, NGRAPH, HID, EMB, 0, stream);
}
